// MultiHeadAttention_30202210026042
// MI455X (gfx1250) — matrix operand order repaired and checked by reading, not compiled, not run on hardware
//
#include <hip/hip_runtime.h>
#include <hip/hip_bf16.h>
#include <math.h>

// ---------------------------------------------------------------------------
// MI455X (gfx1250) multi-head attention forward, bf16 WMMA everywhere.
//   x:[2,2048,1024] f32, Wq/Wk/Wv/Wo:[1024,1024] f32, bo:[1024] f32
//   out:[2,2048,1024] f32
// Pipeline: (1a) Q/K proj GEMM -> bf16 [b,h,t,d]
//           (1b) V proj GEMM   -> bf16 [b,h,d,t] (pre-transposed)
//           (2)  causal flash-attn (async global->LDS tile staging)
//           (3)  out proj GEMM (+bias) -> f32
// ---------------------------------------------------------------------------

typedef __bf16 bf16;
typedef bf16 v16bf __attribute__((ext_vector_type(16)));
typedef bf16 v8bf  __attribute__((ext_vector_type(8)));
typedef float v8f  __attribute__((ext_vector_type(8)));
typedef int  v4i32 __attribute__((__vector_size__(16)));

#define D_IN  1024
#define D_OUT 1024
#define NH    16
#define HD    64
#define SEQ   2048
#define BATCH 2

// ---- CDNA5 async global->LDS staging (ASYNCcnt path), with safe fallback ----
#if defined(__HIP_DEVICE_COMPILE__) && __has_builtin(__builtin_amdgcn_global_load_async_to_lds_b128)
#define HAVE_ASYNC_LDS 1
#else
#define HAVE_ASYNC_LDS 0
#endif

#define GLOBAL_AS __attribute__((address_space(1)))
#define LDS_AS    __attribute__((address_space(3)))

__device__ __forceinline__ void async_copy16(void* lds_dst, const void* gsrc) {
#if HAVE_ASYNC_LDS
  __builtin_amdgcn_global_load_async_to_lds_b128(
      (GLOBAL_AS v4i32*)gsrc, (LDS_AS v4i32*)lds_dst, /*offset=*/0, /*cpol=*/0);
#else
  *(uint4*)lds_dst = *(const uint4*)gsrc;
#endif
}

__device__ __forceinline__ void wait_async_then_barrier() {
#if HAVE_ASYNC_LDS
#if __has_builtin(__builtin_amdgcn_s_wait_asynccnt)
  __builtin_amdgcn_s_wait_asynccnt(0);
#else
  asm volatile("s_wait_asynccnt 0x0" ::: "memory");
#endif
#endif
  __syncthreads();
}

__device__ __forceinline__ v16bf cat8(v8bf lo, v8bf hi) {
  v16bf r;
#pragma unroll
  for (int i = 0; i < 8; ++i) { r[i] = lo[i]; r[i + 8] = hi[i]; }
  return r;
}

__device__ __forceinline__ v8f vzero8() {
  v8f z;
#pragma unroll
  for (int i = 0; i < 8; ++i) z[i] = 0.0f;
  return z;
}

__device__ __forceinline__ v8f wmma_bf16(v16bf a, v16bf b, v8f c) {
  // D = A(16x32 bf16) * B(32x16 bf16) + C(16x16 f32)
  return __builtin_amdgcn_wmma_f32_16x16x32_bf16(
      /*neg_a=*/false, a, /*neg_b=*/false, b,
      /*c_mod=*/(short)0, c, /*reuse_a=*/false, /*reuse_b=*/false);
}

// ---------------------------------------------------------------------------
// Shared GEMM body for the f32-input projections.
// Tile: BM=128 BN=128 BK=32, 256 threads = 8 waves (2 in M x 4 in N),
// each wave computes a 64x32 region = 4x2 WMMA accumulators.
// ---------------------------------------------------------------------------
__device__ __forceinline__ void proj_gemm_body(
    const float* __restrict__ A, const float* __restrict__ W,
    bf16 (*As)[40], bf16 (*Bt)[40],
    int rowBase, int colBase, int tid, v8f acc[4][2]) {
  const int lane = tid & 31;
  const int wid  = tid >> 5;
  const int hl   = lane >> 4;
  const int ln16 = lane & 15;
  const int wm = (wid & 1) * 64;
  const int wn = (wid >> 1) * 32;

  for (int k0 = 0; k0 < D_IN; k0 += 32) {
    // --- stage A (128x32 f32 -> bf16) ---
#pragma unroll
    for (int i = 0; i < 4; ++i) {
      int s  = tid + i * 256;            // 1024 float4 slots
      int r  = s >> 3;
      int c4 = (s & 7) * 4;
      float4 f = *(const float4*)(A + (size_t)(rowBase + r) * D_IN + k0 + c4);
      As[r][c4 + 0] = (bf16)f.x; As[r][c4 + 1] = (bf16)f.y;
      As[r][c4 + 2] = (bf16)f.z; As[r][c4 + 3] = (bf16)f.w;
    }
    // --- stage B transposed (32x128 f32 -> bf16 Bt[n][k]) ---
#pragma unroll
    for (int i = 0; i < 4; ++i) {
      int s  = tid + i * 256;            // 1024 float4 slots
      int kk = s >> 5;
      int c4 = (s & 31) * 4;
      float4 f = *(const float4*)(W + (size_t)(k0 + kk) * D_OUT + colBase + c4);
      Bt[c4 + 0][kk] = (bf16)f.x; Bt[c4 + 1][kk] = (bf16)f.y;
      Bt[c4 + 2][kk] = (bf16)f.z; Bt[c4 + 3][kk] = (bf16)f.w;
    }
    if (k0 + 32 < D_IN) {  // prefetch next A slab -> global_prefetch_b8
      __builtin_prefetch(A + (size_t)(rowBase + (tid >> 1)) * D_IN + k0 + 32, 0, 3);
    }
    __syncthreads();

    // --- fragments + WMMA ---
    v16bf af[4];
#pragma unroll
    for (int mt = 0; mt < 4; ++mt) {
      const bf16* ap = &As[wm + mt * 16 + ln16][hl * 8];   // A: K chunks at khalf, khalf+16
      af[mt] = cat8(*(const v8bf*)ap, *(const v8bf*)(ap + 16));
    }
    v16bf bfr[2];
#pragma unroll
    for (int nt = 0; nt < 2; ++nt) {
      const bf16* bp = &Bt[wn + nt * 16 + ln16][hl * 8];   // B: K chunks at khalf, khalf+16
      bfr[nt] = cat8(*(const v8bf*)bp, *(const v8bf*)(bp + 16));
    }
#pragma unroll
    for (int mt = 0; mt < 4; ++mt)
#pragma unroll
      for (int nt = 0; nt < 2; ++nt)
        acc[mt][nt] = wmma_bf16(af[mt], bfr[nt], acc[mt][nt]);
    __syncthreads();
  }
}

// ---------------------------------------------------------------------------
// Kernel 1a: Q/K projection. z selects {Wq,Wk}; stores [b,h,t,d] bf16.
// ---------------------------------------------------------------------------
__global__ __launch_bounds__(256) void qk_proj_kernel(
    const float* __restrict__ x, const float* __restrict__ Wq,
    const float* __restrict__ Wk,
    bf16* __restrict__ qout, bf16* __restrict__ kout) {
  __shared__ bf16 As[128][40];
  __shared__ bf16 Bt[128][40];

  const int tid  = threadIdx.x;
  const int lane = tid & 31;
  const int wid  = tid >> 5;
  const int hl   = lane >> 4;
  const int ln16 = lane & 15;
  const int rowBase = blockIdx.y * 128;
  const int colBase = blockIdx.x * 128;
  const float* W = (blockIdx.z == 0) ? Wq : Wk;
  bf16* outp     = (blockIdx.z == 0) ? qout : kout;
  const int wm = (wid & 1) * 64;
  const int wn = (wid >> 1) * 32;

  v8f acc[4][2];
#pragma unroll
  for (int mt = 0; mt < 4; ++mt)
#pragma unroll
    for (int nt = 0; nt < 2; ++nt) acc[mt][nt] = vzero8();

  proj_gemm_body(x, W, As, Bt, rowBase, colBase, tid, acc);

#pragma unroll
  for (int mt = 0; mt < 4; ++mt)
#pragma unroll
    for (int nt = 0; nt < 2; ++nt)
#pragma unroll
      for (int r = 0; r < 8; ++r) {
        int row = rowBase + wm + mt * 16 + r + hl * 8;
        int col = colBase + wn + nt * 16 + ln16;
        int bb = row >> 11, t = row & 2047;
        int hh = col >> 6,  d = col & 63;
        outp[(((size_t)(bb * NH + hh)) * SEQ + t) * HD + d] = (bf16)acc[mt][nt][r];
      }
}

// ---------------------------------------------------------------------------
// Kernel 1b: V projection, stored TRANSPOSED [b,h,d,t] bf16 so the attention
// kernel can stage V tiles with contiguous async-to-LDS copies.
// ---------------------------------------------------------------------------
__global__ __launch_bounds__(256) void v_proj_kernel(
    const float* __restrict__ x, const float* __restrict__ Wv,
    bf16* __restrict__ vout) {
  __shared__ bf16 As[128][40];
  __shared__ bf16 Bt[128][40];

  const int tid  = threadIdx.x;
  const int lane = tid & 31;
  const int wid  = tid >> 5;
  const int hl   = lane >> 4;
  const int ln16 = lane & 15;
  const int rowBase = blockIdx.y * 128;
  const int colBase = blockIdx.x * 128;
  const int wm = (wid & 1) * 64;
  const int wn = (wid >> 1) * 32;

  v8f acc[4][2];
#pragma unroll
  for (int mt = 0; mt < 4; ++mt)
#pragma unroll
    for (int nt = 0; nt < 2; ++nt) acc[mt][nt] = vzero8();

  proj_gemm_body(x, Wv, As, Bt, rowBase, colBase, tid, acc);

#pragma unroll
  for (int mt = 0; mt < 4; ++mt)
#pragma unroll
    for (int nt = 0; nt < 2; ++nt)
#pragma unroll
      for (int r = 0; r < 8; ++r) {
        int row = rowBase + wm + mt * 16 + r + hl * 8;
        int col = colBase + wn + nt * 16 + ln16;
        int bb = row >> 11, t = row & 2047;
        int hh = col >> 6,  d = col & 63;
        vout[(((size_t)(bb * NH + hh)) * HD + d) * SEQ + t] = (bf16)acc[mt][nt][r];
      }
}

// ---------------------------------------------------------------------------
// Kernel 2: causal flash-attention (no 1/sqrt(d) scaling, per reference).
// Grid: (32 q-tiles, 16 heads, 2 batches); 128 threads = 4 waves,
// wave w owns q rows [qt*64 + w*16, +16). K-tiles of 64, online softmax.
// K tile and pre-transposed V tile staged with async global->LDS copies.
// ---------------------------------------------------------------------------
__global__ __launch_bounds__(128) void attn_kernel(
    const bf16* __restrict__ Q, const bf16* __restrict__ K,
    const bf16* __restrict__ Vt_g, bf16* __restrict__ ctxout) {
  __shared__ bf16 Ks[64][72];      // K tile, row-major [kc][d]
  __shared__ bf16 Vt[64][72];      // V tile, transposed [d][kc]
  __shared__ bf16 Ps[4][16][72];   // per-wave P tile [qrow][kc]

  const int tid  = threadIdx.x;
  const int lane = tid & 31;
  const int w    = tid >> 5;       // wave id 0..3
  const int hl   = lane >> 4;
  const int ln16 = lane & 15;
  const int qt = blockIdx.x;
  const int h  = blockIdx.y;
  const int b  = blockIdx.z;

  const size_t bh = ((size_t)b * NH + h) * (size_t)SEQ * HD;
  const bf16* Qb  = Q    + bh;
  const bf16* Kb  = K    + bh;
  const bf16* Vtb = Vt_g + bh;     // [d][t] within this (b,h)

  // Q A-fragments for this wave's 16 rows (contraction d = 64 -> 2 K-steps)
  const int qrow = qt * 64 + w * 16 + ln16;
  v16bf qf[2];
#pragma unroll
  for (int ks = 0; ks < 2; ++ks) {
    const bf16* qp = Qb + (size_t)qrow * HD + ks * 32 + hl * 8;
    qf[ks] = cat8(*(const v8bf*)qp, *(const v8bf*)(qp + 16));
  }

  float m[8], l[8];
  v8f ctx[4];
#pragma unroll
  for (int r = 0; r < 8; ++r) { m[r] = -INFINITY; l[r] = 0.0f; }
#pragma unroll
  for (int nt = 0; nt < 4; ++nt) ctx[nt] = vzero8();

  for (int kt = 0; kt <= qt; ++kt) {
    // --- async stage: K tile rows + pre-transposed V tile rows ---
#pragma unroll
    for (int i = 0; i < 4; ++i) {
      int s = tid + i * 128;            // 512 x 16B (64 rows x 8 chunks)
      int r = s >> 3, c8 = (s & 7) * 8;
      async_copy16(&Ks[r][c8], Kb + (size_t)(kt * 64 + r) * HD + c8);
    }
#pragma unroll
    for (int i = 0; i < 4; ++i) {
      int s = tid + i * 128;            // 512 x 16B (64 d-rows x 8 chunks)
      int d = s >> 3, c8 = (s & 7) * 8;
      async_copy16(&Vt[d][c8], Vtb + (size_t)d * SEQ + kt * 64 + c8);
    }
    wait_async_then_barrier();

    // --- S = Q K^T (16 x 64), contraction over d ---
    v8f S[4];
#pragma unroll
    for (int nt = 0; nt < 4; ++nt) {
      S[nt] = vzero8();
#pragma unroll
      for (int ks = 0; ks < 2; ++ks) {
        // B = K^T: B[d][kc] = K[kc][d]; lane = kc col, elems = d chunks at dhalf, dhalf+16
        const bf16* kp = &Ks[nt * 16 + ln16][ks * 32 + hl * 8];
        v16bf kf = cat8(*(const v8bf*)kp, *(const v8bf*)(kp + 16));
        S[nt] = wmma_bf16(qf[ks], kf, S[nt]);
      }
    }

    // --- causal mask (diagonal tile only) ---
    if (kt == qt) {
#pragma unroll
      for (int nt = 0; nt < 4; ++nt) {
        int kc = kt * 64 + nt * 16 + ln16;
#pragma unroll
        for (int r = 0; r < 8; ++r) {
          int qr = qt * 64 + w * 16 + r + hl * 8;
          if (kc > qr) S[nt][r] = -INFINITY;
        }
      }
    }

    // --- online softmax; rows live across a 16-lane half-wave ---
#pragma unroll
    for (int r = 0; r < 8; ++r) {
      float rmax = fmaxf(fmaxf(S[0][r], S[1][r]), fmaxf(S[2][r], S[3][r]));
#pragma unroll
      for (int off = 1; off < 16; off <<= 1)
        rmax = fmaxf(rmax, __shfl_xor(rmax, off, 32));
      float mn    = fmaxf(m[r], rmax);
      float alpha = __expf(m[r] - mn);
      float rsum  = 0.0f;
#pragma unroll
      for (int nt = 0; nt < 4; ++nt) {
        float p = __expf(S[nt][r] - mn);
        rsum += p;
        Ps[w][r + hl * 8][nt * 16 + ln16] = (bf16)p;   // C-layout -> LDS
      }
#pragma unroll
      for (int off = 1; off < 16; off <<= 1)
        rsum += __shfl_xor(rsum, off, 32);
      l[r] = l[r] * alpha + rsum;
      m[r] = mn;
#pragma unroll
      for (int nt = 0; nt < 4; ++nt) ctx[nt][r] *= alpha;
    }
    __syncthreads();   // P visible as A-fragments; Ks reads done

    // --- ctx += P @ V (contraction over kc = 64 -> 2 K-steps) ---
#pragma unroll
    for (int ks = 0; ks < 2; ++ks) {
      const bf16* pp = &Ps[w][ln16][ks * 32 + hl * 8];
      v16bf pf = cat8(*(const v8bf*)pp, *(const v8bf*)(pp + 16));
#pragma unroll
      for (int nt = 0; nt < 4; ++nt) {
        // B = V: lane = d col, elems = kc chunks at khalf, khalf+16 (Vt rows are d)
        const bf16* vp = &Vt[nt * 16 + ln16][ks * 32 + hl * 8];
        v16bf vf = cat8(*(const v8bf*)vp, *(const v8bf*)(vp + 16));
        ctx[nt] = wmma_bf16(pf, vf, ctx[nt]);
      }
    }
    __syncthreads();   // before next tile overwrites Ks/Vt/Ps
  }

  // --- normalize and store ctx bf16 as [b,t,h*64+d] for the out-proj GEMM ---
#pragma unroll
  for (int nt = 0; nt < 4; ++nt)
#pragma unroll
    for (int r = 0; r < 8; ++r) {
      float val = ctx[nt][r] / l[r];
      int t   = qt * 64 + w * 16 + r + hl * 8;
      int col = h * 64 + nt * 16 + ln16;
      ctxout[((size_t)(b * SEQ + t)) * D_OUT + col] = (bf16)val;
    }
}

// ---------------------------------------------------------------------------
// Kernel 3: out = ctx @ Wo + bo (f32 out). bf16 A staged via async-to-LDS.
// ---------------------------------------------------------------------------
__global__ __launch_bounds__(256) void out_proj_kernel(
    const bf16* __restrict__ Cx, const float* __restrict__ Wo,
    const float* __restrict__ bo, float* __restrict__ out) {
  __shared__ bf16 As[128][40];
  __shared__ bf16 Bt[128][40];

  const int tid  = threadIdx.x;
  const int lane = tid & 31;
  const int wid  = tid >> 5;
  const int hl   = lane >> 4;
  const int ln16 = lane & 15;
  const int rowBase = blockIdx.y * 128;
  const int colBase = blockIdx.x * 128;
  const int wm = (wid & 1) * 64;
  const int wn = (wid >> 1) * 32;

  v8f acc[4][2];
#pragma unroll
  for (int mt = 0; mt < 4; ++mt)
#pragma unroll
    for (int nt = 0; nt < 2; ++nt) acc[mt][nt] = vzero8();

  for (int k0 = 0; k0 < D_OUT; k0 += 32) {
    // --- stage A (bf16 row-major) with async global->LDS b128 copies ---
#pragma unroll
    for (int i = 0; i < 2; ++i) {
      int s = tid + i * 256;             // 512 x 16B (128 rows x 4 chunks)
      int r = s >> 2, c8 = (s & 3) * 8;
      async_copy16(&As[r][c8], Cx + (size_t)(rowBase + r) * D_OUT + k0 + c8);
    }
    // --- stage B transposed (32x128 f32 -> bf16 Bt[n][k]) ---
#pragma unroll
    for (int i = 0; i < 4; ++i) {
      int s  = tid + i * 256;
      int kk = s >> 5;
      int c4 = (s & 31) * 4;
      float4 f = *(const float4*)(Wo + (size_t)(k0 + kk) * D_OUT + colBase + c4);
      Bt[c4 + 0][kk] = (bf16)f.x; Bt[c4 + 1][kk] = (bf16)f.y;
      Bt[c4 + 2][kk] = (bf16)f.z; Bt[c4 + 3][kk] = (bf16)f.w;
    }
    wait_async_then_barrier();

    v16bf af[4];
#pragma unroll
    for (int mt = 0; mt < 4; ++mt) {
      const bf16* ap = &As[wm + mt * 16 + ln16][hl * 8];
      af[mt] = cat8(*(const v8bf*)ap, *(const v8bf*)(ap + 16));
    }
    v16bf bfr[2];
#pragma unroll
    for (int nt = 0; nt < 2; ++nt) {
      const bf16* bp = &Bt[wn + nt * 16 + ln16][hl * 8];
      bfr[nt] = cat8(*(const v8bf*)bp, *(const v8bf*)(bp + 16));
    }
#pragma unroll
    for (int mt = 0; mt < 4; ++mt)
#pragma unroll
      for (int nt = 0; nt < 2; ++nt)
        acc[mt][nt] = wmma_bf16(af[mt], bfr[nt], acc[mt][nt]);
    __syncthreads();
  }

#pragma unroll
  for (int mt = 0; mt < 4; ++mt)
#pragma unroll
    for (int nt = 0; nt < 2; ++nt)
#pragma unroll
      for (int r = 0; r < 8; ++r) {
        int row = rowBase + wm + mt * 16 + r + hl * 8;
        int col = colBase + wn + nt * 16 + ln16;
        out[(size_t)row * D_OUT + col] = acc[mt][nt][r] + bo[col];
      }
}

// ---------------------------------------------------------------------------
extern "C" void kernel_launch(void* const* d_in, const int* in_sizes, int n_in,
                              void* d_out, int out_size, void* d_ws, size_t ws_size,
                              hipStream_t stream) {
  const float* x  = (const float*)d_in[0];
  const float* Wq = (const float*)d_in[1];
  const float* Wk = (const float*)d_in[2];
  const float* Wv = (const float*)d_in[3];
  const float* Wo = (const float*)d_in[4];
  const float* bo = (const float*)d_in[5];
  float* out = (float*)d_out;

  char* ws = (char*)d_ws;
  const size_t qkv_bytes = (size_t)BATCH * NH * SEQ * HD * sizeof(bf16); // 8 MB
  bf16* q  = (bf16*)(ws + 0 * qkv_bytes);
  bf16* k  = (bf16*)(ws + 1 * qkv_bytes);
  bf16* vt = (bf16*)(ws + 2 * qkv_bytes);   // V stored [b,h,d,t]
  bf16* c  = (bf16*)(ws + 3 * qkv_bytes);

  dim3 g1(D_OUT / 128, (BATCH * SEQ) / 128, 2);   // 8 x 32 x {Wq,Wk}
  qk_proj_kernel<<<g1, 256, 0, stream>>>(x, Wq, Wk, q, k);

  dim3 g1v(D_OUT / 128, (BATCH * SEQ) / 128);     // 8 x 32 (Wv)
  v_proj_kernel<<<g1v, 256, 0, stream>>>(x, Wv, vt);

  dim3 g2(SEQ / 64, NH, BATCH);                   // 32 x 16 x 2
  attn_kernel<<<g2, 128, 0, stream>>>(q, k, vt, c);

  dim3 g3(D_OUT / 128, (BATCH * SEQ) / 128);      // 8 x 32
  out_proj_kernel<<<g3, 256, 0, stream>>>(c, Wo, bo, out);
}
